// GroupedQueryAttention_68478958567935
// MI455X (gfx1250) — matrix operand order repaired and checked by reading, not compiled, not run on hardware
//
#include <hip/hip_runtime.h>
#include <hip/hip_bf16.h>
#include <math.h>

typedef __attribute__((ext_vector_type(16))) _Float16 v16h;
typedef __attribute__((ext_vector_type(8)))  _Float16 v8h;
typedef __attribute__((ext_vector_type(8)))  float    v8f;
typedef __attribute__((ext_vector_type(4)))  int      v4i;
typedef __attribute__((address_space(1))) v4i* gv4i_p;   // global int4*
typedef __attribute__((address_space(3))) v4i* lv4i_p;   // LDS int4*

#define B_  2
#define T_  2048
#define D_  2048
#define H_  16
#define KV_ 4
#define HD_ 128
#define GQ_ (H_/KV_)

#if __has_builtin(__builtin_amdgcn_global_load_async_to_lds_b128) && \
    __has_builtin(__builtin_amdgcn_s_wait_asynccnt)
#define HAVE_ASYNC_LDS 1
#else
#define HAVE_ASYNC_LDS 0
#endif

// ---------------------------------------------------------------- conversions
__global__ void cvt_f32_f16(const float* __restrict__ in, _Float16* __restrict__ out, int n) {
  int i = blockIdx.x * blockDim.x + threadIdx.x;
  if (i < n) out[i] = (_Float16)in[i];
}

// -------------------------------------------------------------------- GEMM
// C[M,N](f32) = A[M,K](f16) * B[K,N](f16), row-major. M%128==0, N%64==0, K%32==0.
// 256 threads = 8 waves; block tile 128x64; wave tile 32x32 (2x2 WMMA).
// A tile staged row-major (async global->LDS when available);
// B tile staged TRANSPOSED so B-fragments are contiguous ds_load_b128 pairs.
__global__ __launch_bounds__(256) void gemm_f16(const _Float16* __restrict__ A,
                                                const _Float16* __restrict__ Bm,
                                                float* __restrict__ C,
                                                int M, int N, int K) {
  __shared__ _Float16 As[128 * 32];   // [row][k]
  __shared__ _Float16 BsT[64 * 32];   // [col][k]  (transposed)
  const int tid  = threadIdx.x;
  const int lane = tid & 31;
  const int wid  = tid >> 5;
  const int wr   = wid >> 1;        // 0..3 row wave
  const int wc   = wid & 1;         // 0..1 col wave
  const int lx   = lane & 15;
  const int ly   = lane >> 4;
  const int rowBlk = blockIdx.y * 128;
  const int colBlk = blockIdx.x * 64;

  v8f acc[2][2] = {};

  for (int k0 = 0; k0 < K; k0 += 32) {
    { // A tile 128x32 -> LDS
      int r = tid >> 1;
      int c = (tid & 1) * 16;
      const _Float16* src = A + (size_t)(rowBlk + r) * K + k0 + c;
#if HAVE_ASYNC_LDS
      __builtin_amdgcn_global_load_async_to_lds_b128(
          (gv4i_p)src,       (lv4i_p)&As[r * 32 + c],     0, 0);
      __builtin_amdgcn_global_load_async_to_lds_b128(
          (gv4i_p)(src + 8), (lv4i_p)&As[r * 32 + c + 8], 0, 0);
#else
      *(v8h*)&As[r * 32 + c]     = ((const v8h*)src)[0];
      *(v8h*)&As[r * 32 + c + 8] = ((const v8h*)src)[1];
#endif
    }
    { // B tile 32x64 -> LDS transposed: BsT[col][k]
      int r = tid >> 3;          // k row 0..31
      int c = (tid & 7) * 8;     // col 0..56
      v8h bv = *(const v8h*)(Bm + (size_t)(k0 + r) * N + colBlk + c);
      #pragma unroll
      for (int j = 0; j < 8; ++j) BsT[(c + j) * 32 + r] = bv[j];
    }
#if HAVE_ASYNC_LDS
    __builtin_amdgcn_s_wait_asynccnt(0);
#endif
    __syncthreads();

    v16h afrag[2], bfrag[2];
    {
      int kb = ly * 16;
      #pragma unroll
      for (int t = 0; t < 2; ++t) {
        int row = wr * 32 + t * 16 + lx;
        v8h lo = *(const v8h*)&As[row * 32 + kb];
        v8h hi = *(const v8h*)&As[row * 32 + kb + 8];
        v16h a;
        #pragma unroll
        for (int j = 0; j < 8; ++j) { a[j] = lo[j]; a[8 + j] = hi[j]; }
        afrag[t] = a;
      }
      int kbb = ly * 16;
      #pragma unroll
      for (int t = 0; t < 2; ++t) {
        int col = wc * 32 + t * 16 + lx;
        bfrag[t] = *(const v16h*)&BsT[col * 32 + kbb];   // contiguous 32B
      }
    }
    #pragma unroll
    for (int ti = 0; ti < 2; ++ti)
      #pragma unroll
      for (int tj = 0; tj < 2; ++tj)
        acc[ti][tj] = __builtin_amdgcn_wmma_f32_16x16x32_f16(
            false, afrag[ti], false, bfrag[tj], (short)0, acc[ti][tj], false, false);
    __syncthreads();
  }

  #pragma unroll
  for (int ti = 0; ti < 2; ++ti)
    #pragma unroll
    for (int tj = 0; tj < 2; ++tj) {
      int rbase = rowBlk + wr * 32 + ti * 16 + 8 * ly;
      int cidx  = colBlk + wc * 32 + tj * 16 + lx;
      #pragma unroll
      for (int i = 0; i < 8; ++i)
        C[(size_t)(rbase + i) * N + cidx] = acc[ti][tj][i];
    }
}

// -------------------------------------------------------------------- RoPE
// q: f32 (B,T,D) -> f16 (B,H,T,HD) with rope. n = B*T*H*64 = 2^22
__global__ void rope_q(const float* __restrict__ qf, _Float16* __restrict__ qh, int n) {
  int idx = blockIdx.x * blockDim.x + threadIdx.x;
  if (idx >= n) return;
  int i = idx & 63;
  int h = (idx >> 6) & (H_ - 1);
  int t = (idx >> 10) & (T_ - 1);
  int b = idx >> 21;
  const float* row = qf + ((size_t)(b * T_ + t)) * D_ + h * HD_;
  float q1 = row[2 * i], q2 = row[2 * i + 1];
  float inv = __expf(-9.210340371976184f * ((float)i * (1.0f / 64.0f))); // 10000^(-2i/128)
  float s, c;
  __sincosf((float)t * inv, &s, &c);
  size_t ob = (((size_t)b * H_ + h) * T_ + t) * HD_;
  qh[ob + i]      = (_Float16)(q1 * c - q2 * s);
  qh[ob + 64 + i] = (_Float16)(q1 * s + q2 * c);
}

// k: f32 (B,T,KV*HD) -> f16 (B,KV,T,HD) with rope. n = B*T*KV*64 = 2^20
__global__ void rope_k(const float* __restrict__ kf, _Float16* __restrict__ kh, int n) {
  int idx = blockIdx.x * blockDim.x + threadIdx.x;
  if (idx >= n) return;
  int i  = idx & 63;
  int kv = (idx >> 6) & (KV_ - 1);
  int t  = (idx >> 8) & (T_ - 1);
  int b  = idx >> 19;
  const float* row = kf + ((size_t)(b * T_ + t)) * (KV_ * HD_) + kv * HD_;
  float k1 = row[2 * i], k2 = row[2 * i + 1];
  float inv = __expf(-9.210340371976184f * ((float)i * (1.0f / 64.0f)));
  float s, c;
  __sincosf((float)t * inv, &s, &c);
  size_t ob = (((size_t)b * KV_ + kv) * T_ + t) * HD_;
  kh[ob + i]      = (_Float16)(k1 * c - k2 * s);
  kh[ob + 64 + i] = (_Float16)(k1 * s + k2 * c);
}

// v: f32 (B,T,KV*HD) -> f16 TRANSPOSED (B,KV,HD,T) so attention B-fragments
// (K-dim contiguous) are single 32B loads. n = B*T*KV*HD = 2^21
__global__ void cvt_v(const float* __restrict__ vf, _Float16* __restrict__ vt, int n) {
  int idx = blockIdx.x * blockDim.x + threadIdx.x;
  if (idx >= n) return;
  int d  = idx & 127;
  int kv = (idx >> 7) & (KV_ - 1);
  int t  = (idx >> 9) & (T_ - 1);
  int b  = idx >> 20;
  vt[(((size_t)b * KV_ + kv) * HD_ + d) * T_ + t] =
      (_Float16)vf[((size_t)(b * T_ + t)) * (KV_ * HD_) + kv * HD_ + d];
}

// ------------------------------------------------------------- attention
// One wave handles 16 query rows; flash-style online softmax; all matmuls WMMA.
// grid: (T/128, B*H), block 256 (8 independent waves).
__global__ __launch_bounds__(256) void attn_kernel(const _Float16* __restrict__ qh,
                                                   const _Float16* __restrict__ kh,
                                                   const _Float16* __restrict__ vt,
                                                   _Float16* __restrict__ oh) {
  __shared__ float Pl[8 * 16 * 32];   // per-wave 16x32 P transpose scratch
  const int tid  = threadIdx.x;
  const int lane = tid & 31;
  const int wid  = tid >> 5;
  const int lx   = lane & 15;
  const int ly   = lane >> 4;
  const int q0   = blockIdx.x * 128 + wid * 16;
  const int bh   = blockIdx.y;
  const int b    = bh / H_, h = bh % H_, kv = h / GQ_;
  const _Float16* Q  = qh + (((size_t)b * H_ + h) * T_) * HD_;
  const _Float16* Kp = kh + (((size_t)b * KV_ + kv) * T_) * HD_;
  const _Float16* Vt = vt + (((size_t)b * KV_ + kv) * HD_) * T_;   // [hd][t]
  float* pl = &Pl[wid * 512];

  // Q fragments: 4 chunks of K=32 over HD=128 (A-matrix layout)
  v16h qfrag[4];
  {
    int row = q0 + lx;
    int kb  = ly * 16;
    #pragma unroll
    for (int c = 0; c < 4; ++c) {
      v8h lo = *(const v8h*)(Q + (size_t)row * HD_ + c * 32 + kb);
      v8h hi = *(const v8h*)(Q + (size_t)row * HD_ + c * 32 + kb + 8);
      v16h a;
      #pragma unroll
      for (int j = 0; j < 8; ++j) { a[j] = lo[j]; a[8 + j] = hi[j]; }
      qfrag[c] = a;
    }
  }

  v8f o[8] = {};
  float m[8], l[8];
  #pragma unroll
  for (int i = 0; i < 8; ++i) { m[i] = -3.0e38f; l[i] = 0.f; }
  const float scale = 0.08838834764831845f; // 1/sqrt(128)

  for (int k0 = 0; k0 <= q0 + 15; k0 += 32) {
    // prefetch next K/V chunk (global_prefetch_b8); K/V live in L2 anyway
    if (k0 + 32 <= q0 + 15) {
      __builtin_prefetch((const void*)(Kp + (size_t)(k0 + 32 + lx) * HD_), 0, 1);
      __builtin_prefetch((const void*)(Vt + (size_t)(lane * 4) * T_ + k0 + 32), 0, 1);
    }
    // S = Q K^T : 16x32 as two 16x16 accumulators
    v8f s0 = {}, s1 = {};
    #pragma unroll
    for (int c = 0; c < 4; ++c) {
      v16h b0 = *(const v16h*)(Kp + (size_t)(k0 + lx) * HD_ + c * 32 + ly * 16);
      v16h b1 = *(const v16h*)(Kp + (size_t)(k0 + 16 + lx) * HD_ + c * 32 + ly * 16);
      s0 = __builtin_amdgcn_wmma_f32_16x16x32_f16(false, qfrag[c], false, b0, (short)0, s0, false, false);
      s1 = __builtin_amdgcn_wmma_f32_16x16x32_f16(false, qfrag[c], false, b1, (short)0, s1, false, false);
    }
    const bool needMask = (k0 + 31 > q0);
    float rmax[8];
    #pragma unroll
    for (int i = 0; i < 8; ++i) {
      float a0 = s0[i] * scale, a1 = s1[i] * scale;
      if (needMask) {
        int rrow = q0 + i + 8 * ly;
        if (k0 + lx > rrow)      a0 = -3.0e38f;
        if (k0 + 16 + lx > rrow) a1 = -3.0e38f;
      }
      s0[i] = a0; s1[i] = a1;
      rmax[i] = fmaxf(a0, a1);
    }
    // row reductions stay within each 16-lane half (matches C layout)
    #pragma unroll
    for (int i = 0; i < 8; ++i) {
      float r = rmax[i];
      r = fmaxf(r, __shfl_xor(r, 1, 32));
      r = fmaxf(r, __shfl_xor(r, 2, 32));
      r = fmaxf(r, __shfl_xor(r, 4, 32));
      r = fmaxf(r, __shfl_xor(r, 8, 32));
      rmax[i] = r;
    }
    float alpha[8];
    #pragma unroll
    for (int i = 0; i < 8; ++i) {
      float mn = fmaxf(m[i], rmax[i]);
      alpha[i] = __expf(m[i] - mn);
      m[i] = mn;
      float p0 = __expf(s0[i] - mn);
      float p1 = __expf(s1[i] - mn);
      s0[i] = p0; s1[i] = p1;
      float r = p0 + p1;
      r += __shfl_xor(r, 1, 32);
      r += __shfl_xor(r, 2, 32);
      r += __shfl_xor(r, 4, 32);
      r += __shfl_xor(r, 8, 32);
      l[i] = l[i] * alpha[i] + r;
    }
    #pragma unroll
    for (int n = 0; n < 8; ++n)
      #pragma unroll
      for (int i = 0; i < 8; ++i)
        o[n][i] *= alpha[i];

    // P (C layout) -> LDS -> A-fragment layout (in-wave transpose)
    #pragma unroll
    for (int i = 0; i < 8; ++i) {
      int rr = i + 8 * ly;
      pl[rr * 32 + lx]      = s0[i];
      pl[rr * 32 + 16 + lx] = s1[i];
    }
#if __has_builtin(__builtin_amdgcn_s_wait_dscnt)
    __builtin_amdgcn_s_wait_dscnt(0);
    asm volatile("" ::: "memory");
#else
    asm volatile("s_wait_dscnt 0" ::: "memory");
#endif
    v16h pa;
    {
      int kb = ly * 16;
      #pragma unroll
      for (int j = 0; j < 8; ++j) {
        pa[j]     = (_Float16)pl[lx * 32 + kb + j];
        pa[8 + j] = (_Float16)pl[lx * 32 + kb + 8 + j];
      }
    }
    // O += P * V   (V^T layout: K-dim contiguous, one 32B load per fragment)
    #pragma unroll
    for (int n = 0; n < 8; ++n) {
      v16h vfr = *(const v16h*)(Vt + (size_t)(n * 16 + lx) * T_ + k0 + ly * 16);
      o[n] = __builtin_amdgcn_wmma_f32_16x16x32_f16(false, pa, false, vfr, (short)0, o[n], false, false);
    }
  }

  #pragma unroll
  for (int i = 0; i < 8; ++i) l[i] = 1.0f / l[i];
  #pragma unroll
  for (int n = 0; n < 8; ++n)
    #pragma unroll
    for (int i = 0; i < 8; ++i) {
      int rr = q0 + i + 8 * ly;
      oh[((size_t)(b * T_ + rr)) * D_ + h * HD_ + n * 16 + lx] = (_Float16)(o[n][i] * l[i]);
    }
}

// ------------------------------------------------------------------ launch
extern "C" void kernel_launch(void* const* d_in, const int* in_sizes, int n_in,
                              void* d_out, int out_size, void* d_ws, size_t ws_size,
                              hipStream_t stream) {
  (void)in_sizes; (void)n_in; (void)out_size; (void)ws_size;
  const float* x  = (const float*)d_in[0];
  const float* Wq = (const float*)d_in[1];
  const float* Wk = (const float*)d_in[2];
  const float* Wv = (const float*)d_in[3];
  const float* Wo = (const float*)d_in[4];
  float* out = (float*)d_out;

  const size_t n_x  = (size_t)B_ * T_ * D_;       // 8388608
  const size_t n_Wq = (size_t)D_ * D_;            // 4194304
  const size_t n_Wk = (size_t)D_ * KV_ * HD_;     // 1048576
  const size_t n_kv = (size_t)B_ * T_ * KV_ * HD_;// 2097152

  char* p = (char*)d_ws;
  auto carve = [&](size_t bytes) { char* r = p; p += (bytes + 255) & ~(size_t)255; return r; };
  _Float16* xh   = (_Float16*)carve(n_x  * 2);
  _Float16* Wqh  = (_Float16*)carve(n_Wq * 2);
  _Float16* Wkh  = (_Float16*)carve(n_Wk * 2);
  _Float16* Wvh  = (_Float16*)carve(n_Wk * 2);
  _Float16* Woh  = (_Float16*)carve(n_Wq * 2);
  float*    qf   = (float*)   carve(n_x  * 4);
  float*    kf   = (float*)   carve(n_kv * 4);
  float*    vf   = (float*)   carve(n_kv * 4);
  _Float16* qhh  = (_Float16*)carve(n_x  * 2);
  _Float16* khh  = (_Float16*)carve(n_kv * 2);
  _Float16* vth  = (_Float16*)carve(n_kv * 2);
  _Float16* atth = (_Float16*)carve(n_x  * 2);

  auto cdiv = [](size_t a, size_t b) { return (unsigned)((a + b - 1) / b); };

  cvt_f32_f16<<<cdiv(n_x, 256),  256, 0, stream>>>(x,  xh,  (int)n_x);
  cvt_f32_f16<<<cdiv(n_Wq, 256), 256, 0, stream>>>(Wq, Wqh, (int)n_Wq);
  cvt_f32_f16<<<cdiv(n_Wk, 256), 256, 0, stream>>>(Wk, Wkh, (int)n_Wk);
  cvt_f32_f16<<<cdiv(n_Wk, 256), 256, 0, stream>>>(Wv, Wvh, (int)n_Wk);
  cvt_f32_f16<<<cdiv(n_Wq, 256), 256, 0, stream>>>(Wo, Woh, (int)n_Wq);

  const int M = B_ * T_;                 // 4096
  // projections
  gemm_f16<<<dim3(D_ / 64, M / 128), 256, 0, stream>>>(xh, Wqh, qf, M, D_, D_);
  gemm_f16<<<dim3((KV_ * HD_) / 64, M / 128), 256, 0, stream>>>(xh, Wkh, kf, M, KV_ * HD_, D_);
  gemm_f16<<<dim3((KV_ * HD_) / 64, M / 128), 256, 0, stream>>>(xh, Wvh, vf, M, KV_ * HD_, D_);

  rope_q<<<cdiv((size_t)B_ * T_ * H_ * 64, 256), 256, 0, stream>>>(qf, qhh, B_ * T_ * H_ * 64);
  rope_k<<<cdiv((size_t)B_ * T_ * KV_ * 64, 256), 256, 0, stream>>>(kf, khh, B_ * T_ * KV_ * 64);
  cvt_v <<<cdiv(n_kv, 256), 256, 0, stream>>>(vf, vth, (int)n_kv);

  attn_kernel<<<dim3(T_ / 128, B_ * H_), 256, 0, stream>>>(qhh, khh, vth, atth);

  // output projection -> f32 d_out
  gemm_f16<<<dim3(D_ / 64, M / 128), 256, 0, stream>>>(atth, Woh, out, M, D_, D_);
}
